// AdaptiveMetaLearnerV1_24309514895365
// MI455X (gfx1250) — matrix operand order repaired and checked by reading, not compiled, not run on hardware
//
#include <hip/hip_runtime.h>

// ---------------------------------------------------------------------------
// AdaptiveMetaLearnerV1 for MI455X (gfx1250, wave32, WMMA + async-to-LDS).
//
// Roofline: ~336 MB of state-tensor traffic dominates (hx/cx/act_hx/act_cx,
// 84 MB each) -> ~14.5 us at 23.3 TB/s.  Compute is ~27 GFLOP; batched into
// v_wmma_f32_16x16x32_bf16 (16 coords per wave-tile, K=40 padded to 64,
// N=160 in ten 16-wide tiles) it hides under the memory stream.  B fragments
// are pre-swizzled once per block into LDS in the exact wave32 layout so the
// inner loop is ds_load_b128 + v_wmma.  hx tiles stream to LDS with
// global_load_async_to_lds_b128 (ASYNCcnt, double buffered across the two
// LSTM layers); cx / next-tile streams are prefetched (global_prefetch).
// ---------------------------------------------------------------------------

typedef float  v8f   __attribute__((ext_vector_type(8)));
typedef __bf16 v16bf __attribute__((ext_vector_type(16)));

constexpr int kB   = 64;
constexpr int kP   = 4096;
constexpr int kH   = 40;       // hidden
constexpr int kG   = 160;      // 4*H gates
constexpr int kL   = 2;
constexpr int kBP  = kB * kP;          // 262144 coordinates
constexpr int kNT  = 10;               // N tiles of 16 over 160
constexpr int kTiles = kBP / 16;       // 16384 tiles per branch
constexpr int kBlocksPerBranch = 512;  // 4 waves each -> 2048 waves/branch
constexpr int kWavesPerBranch  = kBlocksPerBranch * 4;

// LDS layout (bytes)
constexpr int kFragBytes   = 80 * 32 * 32;             // 80 frags * 32 lanes * 32B
constexpr int kParamFloats = 2200;
constexpr int kParamBytes  = kParamFloats * 4;         // 8800
// per-wave: gates[16][160] | abuf0[16][40] | abuf1[16][40] | hbuf[16][40] | stats
constexpr int kWaveFloats  = 16 * kG + 3 * 640 + 48;   // 4528
constexpr int kWaveBytes   = kWaveFloats * 4;          // 18112
constexpr int kSmemBytes   = kFragBytes + kParamBytes + 4 * kWaveBytes; // 163168

struct BranchParams {
  const float* hx;  const float* cx;
  const float* W1;  const float* b1;  const float* Wout; const float* bout;
  const float* Wih; const float* Whh; const float* bih;  const float* bhh;
  const float* g_ih; const float* b_ih; const float* g_hh; const float* b_hh;
  const float* g_c;  const float* b_c;
};

struct KArgs {
  const float* x;
  const float* lam;
  BranchParams bp[2];
  float* out;   // [B*P] x_out
  float* ws;    // [kTiles] per-tile q partials (branch 1)
};

__device__ __forceinline__ float sigm(float x) { return 1.0f / (1.0f + __expf(-x)); }

// Async DMA: 16B per lane, global -> LDS, tracked by ASYNCcnt.
// LDS aperture flat addresses carry the LDS byte offset in addr[31:0].
__device__ __forceinline__ void async_copy_b128(float* lds_dst, const float* gsrc) {
  const uint32_t loff = (uint32_t)(uintptr_t)lds_dst;
  asm volatile("global_load_async_to_lds_b128 %0, %1, off"
               :: "v"(loff), "v"(gsrc)
               : "memory");
}

// Build a 16x32 bf16 A-fragment (ISA 7.12.2 16-bit A layout) from an LDS
// row-major [16][40] f32 tile.  K >= 40 is zero padding.
__device__ __forceinline__ v16bf make_afrag(const float* src, int kc, int lane) {
  const int m = lane & 15, half = lane >> 4;
  v16bf a;
#pragma unroll
  for (int e = 0; e < 16; ++e) {
    const int k = kc * 32 + ((e < 8) ? e : e + 8) + half * 8;
    float v = 0.0f;
    if (k < kH) v = src[m * kH + k];
    a[e] = (__bf16)v;
  }
  return a;
}

// One GEMM + LayerNorm pass: gates (+)= LN(A @ W^T + bias) * gamma + beta.
template <bool ADD>
__device__ __forceinline__ void gemm_ln_pass(
    const __bf16* __restrict__ frag20, const float* __restrict__ aSrc,
    float* __restrict__ gates, const float* __restrict__ bias,
    const float* __restrict__ gamma, const float* __restrict__ beta, int lane) {
  const int col = lane & 15, half = lane >> 4;
  const v16bf a0 = make_afrag(aSrc, 0, lane);
  const v16bf a1 = make_afrag(aSrc, 1, lane);

  v8f acc[kNT];
#pragma unroll
  for (int n = 0; n < kNT; ++n) {
    v8f c = {0.f, 0.f, 0.f, 0.f, 0.f, 0.f, 0.f, 0.f};
    const v16bf b0 = *(const v16bf*)(frag20 + ((n * 2 + 0) << 9) + (lane << 4));
    c = __builtin_amdgcn_wmma_f32_16x16x32_bf16(false, a0, false, b0, (short)0, c, false, false);
    const v16bf b1 = *(const v16bf*)(frag20 + ((n * 2 + 1) << 9) + (lane << 4));
    c = __builtin_amdgcn_wmma_f32_16x16x32_bf16(false, a1, false, b1, (short)0, c, false, false);
    acc[n] = c;
  }

  // bias (pre-LN) -- depends only on the gate column this lane owns
#pragma unroll
  for (int n = 0; n < kNT; ++n) {
    const float bv = bias[n * 16 + col];
#pragma unroll
    for (int j = 0; j < 8; ++j) acc[n][j] += bv;
  }

  // LN stats per row: a row's 160 values live in 16 lanes of one half-wave
  // (C/D layout: rows 0-7 -> lanes 0-15, rows 8-15 -> lanes 16-31).
  float mean[8], rstd[8];
#pragma unroll
  for (int j = 0; j < 8; ++j) {
    float s = 0.f, q = 0.f;
#pragma unroll
    for (int n = 0; n < kNT; ++n) { const float v = acc[n][j]; s += v; q += v * v; }
#pragma unroll
    for (int off = 1; off < 16; off <<= 1) {
      s += __shfl_xor(s, off, 32);
      q += __shfl_xor(q, off, 32);
    }
    const float mu = s * (1.0f / kG);
    mean[j] = mu;
    rstd[j] = rsqrtf(q * (1.0f / kG) - mu * mu + 1e-5f);
  }

#pragma unroll
  for (int n = 0; n < kNT; ++n) {
    const int gcol = n * 16 + col;
    const float gm = gamma[gcol], bt = beta[gcol];
#pragma unroll
    for (int j = 0; j < 8; ++j) {
      const int row = j + 8 * half;
      const float v = (acc[n][j] - mean[j]) * rstd[j] * gm + bt;
      const int idx = row * kG + gcol;
      if (ADD) gates[idx] += v; else gates[idx] = v;
    }
  }
}

__global__ __launch_bounds__(128) void meta_lstm_kernel(KArgs A) {
  extern __shared__ char smem[];
  __bf16* frags = (__bf16*)smem;                       // swizzled B frags
  float*  par   = (float*)(smem + kFragBytes);         // small params
  char*   wbase = smem + kFragBytes + kParamBytes;     // per-wave scratch

  const int tid = (int)threadIdx.x;
  const int br  = (int)(blockIdx.x >> 9);              // 0..511 -> x branch, 512..1023 -> q branch
  const BranchParams& p = A.bp[br];

  // ---- Stage B fragments: W^T chunks in exact wave32 fragment layout ----
  // frag id f = ((layer*2 + mat)*10 + n)*2 + kc ; per lane 16 bf16 (32B).
  for (int idx = tid; idx < 80 * 512; idx += 128) {
    const int e    = idx & 15;
    const int lane = (idx >> 4) & 31;
    const int f    = idx >> 9;
    const int kc   = f & 1;
    const int t1   = f >> 1;
    const int n    = t1 % kNT;
    const int t2   = t1 / kNT;
    const int mat  = t2 & 1;
    const int lyr  = t2 >> 1;
    const int col = lane & 15, half = lane >> 4;
    const int N = n * 16 + col;
    const int K = kc * 32 + half * 8 + ((e < 8) ? e : e + 8); // B 32x16: lanes 0-15 K=0..7,16..23; 16-31 K=8..15,24..31
    const float* W = mat ? p.Whh : p.Wih;
    float v = 0.0f;
    if (K < kH) v = W[(lyr * kG + N) * kH + K];
    frags[idx] = (__bf16)v;
  }
  // ---- Stage small params ----
  // [0]bih [320]bhh [640]g_ih [960]b_ih [1280]g_hh [1600]b_hh
  // [1920]g_c [2000]b_c [2080]W1 [2120]b1 [2160]Wout   (floats)
  for (int i = tid; i < kL * kG; i += 128) {
    par[i]        = p.bih[i];  par[320 + i]  = p.bhh[i];
    par[640 + i]  = p.g_ih[i]; par[960 + i]  = p.b_ih[i];
    par[1280 + i] = p.g_hh[i]; par[1600 + i] = p.b_hh[i];
  }
  for (int i = tid; i < kL * kH; i += 128) { par[1920 + i] = p.g_c[i]; par[2000 + i] = p.b_c[i]; }
  if (tid < kH) { par[2080 + tid] = p.W1[tid]; par[2120 + tid] = p.b1[tid]; par[2160 + tid] = p.Wout[tid]; }
  __syncthreads();

  const int w = tid >> 5, lane = tid & 31;
  float* gates = (float*)(wbase + w * kWaveBytes);   // [16][160]; cols 0..39 reused as c after cell
  float* abuf0 = gates + 16 * kG;                    // [16][40] hx tile, layer 0
  float* abuf1 = abuf0 + 640;                        // [16][40] hx tile, layer 1
  float* hbuf  = abuf1 + 640;                        // [16][40] xt / h
  float* wst   = hbuf + 640;                         // [32] row stats (mean,rstd)
  float* xtile = wst + 32;                           // [16] x values

  const float boutv = p.bout[0];
  const float lam   = A.lam[0];
  const int gw = ((int)blockIdx.x & 511) * 4 + w;

  for (int t = gw; t < kTiles; t += kWavesPerBranch) {
    const int c0 = t << 4;                           // 16 consecutive coordinates
    const float* hxt0 = p.hx + (size_t)c0 * kH;
    const float* hxt1 = p.hx + (size_t)(kBP + c0) * kH;
    const float* cxt0 = p.cx + (size_t)c0 * kH;
    const float* cxt1 = p.cx + (size_t)(kBP + c0) * kH;

    // Kick off both layers' hx tiles as async DMA (2560B each, 16B/lane x5).
#pragma unroll
    for (int i = 0; i < 5; ++i)
      async_copy_b128(abuf0 + i * 128 + lane * 4, hxt0 + i * 128 + lane * 4);
#pragma unroll
    for (int i = 0; i < 5; ++i)
      async_copy_b128(abuf1 + i * 128 + lane * 4, hxt1 + i * 128 + lane * 4);

    // Prefetch this tile's cell states and next tile's streams (128B/lane).
    __builtin_prefetch((const char*)cxt0 + lane * 128, 0, 1);
    __builtin_prefetch((const char*)cxt1 + lane * 128, 0, 1);
    if (t + kWavesPerBranch < kTiles) {
      const size_t cn0 = (size_t)((t + kWavesPerBranch) << 4) * kH;
      __builtin_prefetch((const char*)(p.hx + cn0) + lane * 128, 0, 1);
      __builtin_prefetch((const char*)(p.hx + (size_t)kBP * kH + cn0) + lane * 128, 0, 1);
      __builtin_prefetch((const char*)(p.cx + cn0) + lane * 128, 0, 1);
      __builtin_prefetch((const char*)(p.cx + (size_t)kBP * kH + cn0) + lane * 128, 0, 1);
    }

    if (lane < 16) xtile[lane] = A.x[c0 + lane];

    // xt0 = x * W1 + b1  (rank-1, elementwise)
#pragma unroll
    for (int i = 0; i < 20; ++i) {
      const int e = i * 32 + lane;
      const int m = e / kH, hh = e % kH;
      hbuf[e] = xtile[m] * par[2080 + hh] + par[2120 + hh];
    }

#pragma unroll 1
    for (int l = 0; l < kL; ++l) {
      // Async loads complete in order: <=5 releases layer 0's tile.
      if (l == 0) asm volatile("s_wait_asynccnt 0x5" ::: "memory");
      else        asm volatile("s_wait_asynccnt 0x0" ::: "memory");
      const float* abufl = (l == 0) ? abuf0 : abuf1;
      const float* cxt   = (l == 0) ? cxt0  : cxt1;

      const __bf16* fih = frags + (size_t)((l * 2 + 0) * 20) * 512;
      const __bf16* fhh = frags + (size_t)((l * 2 + 1) * 20) * 512;
      gemm_ln_pass<false>(fih, hbuf,  gates, par + l * kG,       par + 640  + l * kG, par + 960  + l * kG, lane);
      gemm_ln_pass<true >(fhh, abufl, gates, par + 320 + l * kG, par + 1280 + l * kG, par + 1600 + l * kG, lane);

      // LSTM cell (forget-gate bias = 1.0); c overwrites the dead i-gate slot.
#pragma unroll
      for (int i = 0; i < 20; ++i) {
        const int e = i * 32 + lane;
        const int m = e / kH, hh = e % kH;
        const float iv = gates[m * kG + hh];
        const float fv = gates[m * kG + 40 + hh];
        const float gv = gates[m * kG + 80 + hh];
        const float cn = sigm(fv + 1.0f) * cxt[e] + sigm(iv) * tanhf(gv);
        gates[m * kG + hh] = cn;
      }
      // LN(c) stats: two lanes per row
      {
        const int m = lane >> 1, hf = lane & 1;
        float s = 0.f, q = 0.f;
#pragma unroll
        for (int k = 0; k < 20; ++k) { const float v = gates[m * kG + hf * 20 + k]; s += v; q += v * v; }
        s += __shfl_xor(s, 1, 32);
        q += __shfl_xor(q, 1, 32);
        const float mu = s * (1.0f / kH);
        const float rs = rsqrtf(q * (1.0f / kH) - mu * mu + 1e-5f);
        if (hf == 0) { wst[2 * m] = mu; wst[2 * m + 1] = rs; }
      }
      // h = sigmoid(o) * tanh(LN(c))
#pragma unroll
      for (int i = 0; i < 20; ++i) {
        const int e = i * 32 + lane;
        const int m = e / kH, hh = e % kH;
        const float ov = gates[m * kG + 120 + hh];
        const float cl = (gates[m * kG + hh] - wst[2 * m]) * wst[2 * m + 1] * par[1920 + l * kH + hh]
                         + par[2000 + l * kH + hh];
        hbuf[e] = sigm(ov) * tanhf(cl);
      }
    }

    // out = h @ Wout^T + bout  (two lanes per row)
    {
      const int m = lane >> 1, hf = lane & 1;
      float acc = 0.f;
#pragma unroll
      for (int k = 0; k < 20; ++k) { const int hh = hf * 20 + k; acc += hbuf[m * kH + hh] * par[2160 + hh]; }
      acc += __shfl_xor(acc, 1, 32);
      const float val = acc + boutv;
      if (br == 0) {
        if (hf == 0) A.out[c0 + m] = val;
      } else {
        float tv = (hf == 0) ? lam * tanhf(val) : 0.0f;
#pragma unroll
        for (int off = 1; off < 32; off <<= 1) tv += __shfl_xor(tv, off, 32);
        if (lane == 0) A.ws[t] = tv;   // per-tile partial (deterministic)
      }
    }
  }
}

// qt_out[b] = mean over P of lambda*tanh(q) = (sum of 256 tile partials)/4096
__global__ __launch_bounds__(256) void finalize_qt(const float* __restrict__ ws,
                                                   float* __restrict__ qt) {
  __shared__ float red[256];
  const int b = (int)blockIdx.x, tid = (int)threadIdx.x;
  red[tid] = ws[b * 256 + tid];
  __syncthreads();
#pragma unroll
  for (int s = 128; s > 0; s >>= 1) {
    if (tid < s) red[tid] += red[tid + s];
    __syncthreads();
  }
  if (tid == 0) qt[b] = red[0] * (1.0f / (float)kP);
}

extern "C" void kernel_launch(void* const* d_in, const int* in_sizes, int n_in,
                              void* d_out, int out_size, void* d_ws, size_t ws_size,
                              hipStream_t stream) {
  (void)in_sizes; (void)n_in; (void)out_size; (void)ws_size;
  KArgs A;
  A.x   = (const float*)d_in[0];
  A.lam = (const float*)d_in[5];

  auto fill = [&](BranchParams& bp, int b) {
    bp.W1   = (const float*)d_in[b + 0];  bp.b1   = (const float*)d_in[b + 1];
    bp.Wout = (const float*)d_in[b + 2];  bp.bout = (const float*)d_in[b + 3];
    bp.Wih  = (const float*)d_in[b + 4];  bp.Whh  = (const float*)d_in[b + 5];
    bp.bih  = (const float*)d_in[b + 6];  bp.bhh  = (const float*)d_in[b + 7];
    bp.g_ih = (const float*)d_in[b + 8];  bp.b_ih = (const float*)d_in[b + 9];
    bp.g_hh = (const float*)d_in[b + 10]; bp.b_hh = (const float*)d_in[b + 11];
    bp.g_c  = (const float*)d_in[b + 12]; bp.b_c  = (const float*)d_in[b + 13];
  };
  fill(A.bp[0], 6);
  fill(A.bp[1], 20);
  A.bp[0].hx = (const float*)d_in[1];
  A.bp[0].cx = (const float*)d_in[2];
  A.bp[1].hx = (const float*)d_in[3];   // act_hx
  A.bp[1].cx = (const float*)d_in[4];   // act_cx
  A.out = (float*)d_out;
  A.ws  = (float*)d_ws;

  meta_lstm_kernel<<<2 * kBlocksPerBranch, 128, kSmemBytes, stream>>>(A);
  finalize_qt<<<kB, 256, 0, stream>>>((const float*)d_ws, (float*)d_out + kBP);
}
